// InterpretableMultiHeadAttention_14396730376401
// MI455X (gfx1250) — matrix operand order repaired and checked by reading, not compiled, not run on hardware
//
#include <hip/hip_runtime.h>
#include <hip/hip_bf16.h>

// ---------------------------------------------------------------------------
// InterpretableMultiHeadAttention on MI455X (gfx1250), wave32 + bf16 WMMA.
//
// Pipeline (all matrix math on v_wmma_f32_16x16x32_bf16, f32 accumulate):
//   1) convert x -> bf16; convert+transpose Wq/Wk/Wv/Wo -> bf16 [N][K]
//   2) proj GEMM: q,k (per head) and v  ([4096x1024]x[1024x64], +bias)
//   3) attention: per (b, 16-row tile), loop 16 heads in-block:
//        S = q k^T * scale (WMMA), causal mask, f32 softmax over LDS strip,
//        accumulate mean over heads directly into d_out mean_weights,
//        emit bf16 copy of mean_weights on last head.
//      (mean_head == mean_weights @ v because V is shared across heads)
//   4) mh = mean_weights @ v   (causal K range only)
//   5) out = mh @ Wo + bo
//
// Tile staging uses GLOBAL_LOAD_ASYNC_TO_LDS_B128 (ASYNCcnt-tracked
// memory->LDS path) when the toolchain exposes it, else b128 copies.
// ---------------------------------------------------------------------------

constexpr int B_  = 2;
constexpr int T_  = 2048;
constexpr int H_  = 1024;
constexpr int NH_ = 16;
constexpr int D_  = 64;
constexpr int M_  = B_ * T_;          // 4096 flattened (b,t) rows
constexpr float SCALE_ = 0.125f;      // D^-0.5

typedef __attribute__((ext_vector_type(16))) __bf16 v16bf;
typedef __attribute__((ext_vector_type(8)))  float  v8f;
typedef int gv4i __attribute__((vector_size(16)));   // int4 as builtin expects

#define WMMA_BF16(a, b, c) \
  __builtin_amdgcn_wmma_f32_16x16x32_bf16(false, (a), false, (b), (short)0, (c), false, false)

#if defined(__gfx1250__) && __has_builtin(__builtin_amdgcn_global_load_async_to_lds_b128) && \
    __has_builtin(__builtin_amdgcn_s_wait_asynccnt)
#define HAS_ASYNC_LDS 1
#else
#define HAS_ASYNC_LDS 0
#endif

// 16-byte tile-stage copy: global -> LDS.
static __device__ __forceinline__ void cp16(unsigned short* dst, const unsigned short* src) {
#if HAS_ASYNC_LDS
  __builtin_amdgcn_global_load_async_to_lds_b128(
      (__attribute__((address_space(1))) gv4i*)src,
      (__attribute__((address_space(3))) gv4i*)dst, 0, 0);
#else
  *(uint4*)dst = *(const uint4*)src;
#endif
}

// Wait for this wave's outstanding async LDS transfers (no-op on fallback).
static __device__ __forceinline__ void async_join() {
#if HAS_ASYNC_LDS
  __builtin_amdgcn_s_wait_asynccnt(0);
#endif
}

static __device__ __forceinline__ unsigned short f2bf(float x) {
  unsigned u = __builtin_bit_cast(unsigned, x);
  u += 0x7FFFu + ((u >> 16) & 1u);               // round-to-nearest-even
  return (unsigned short)(u >> 16);
}

// A fragment (16x32 bf16) from row-major LDS tile [16][lda].
// A layout (matches frag_b): row = lane&15; lane<16 -> K {0..15},
// lane>=16 -> K {16..31}.  Contiguous dwords -> ds_load_b128.
static __device__ __forceinline__ v16bf frag_a(const unsigned short* sm, int lda,
                                               int lane, int k0) {
  const unsigned short* base = sm + (lane & 15) * lda + k0 + ((lane >> 4) << 4);
  union { unsigned u[8]; v16bf v; } f;
#pragma unroll
  for (int p = 0; p < 8; ++p) {
    int off = p << 1;
    f.u[p] = *(const unsigned*)(base + off);     // 2 bf16 per dword
  }
  return f.v;
}

// B fragment (32x16 bf16) from LDS tile stored transposed: [N][ldk] (K contiguous).
// Lane n = lane&15 selects column; lanes 16-31 cover K+16.
static __device__ __forceinline__ v16bf frag_b(const unsigned short* sm, int ldk,
                                               int lane, int n0, int k0) {
  const unsigned short* base = sm + (size_t)(n0 + (lane & 15)) * ldk + k0 + ((lane >> 4) << 4);
  union { unsigned u[8]; v16bf v; } f;
#pragma unroll
  for (int p = 0; p < 8; ++p) f.u[p] = *(const unsigned*)(base + (p << 1));
  return f.v;
}

// ---------------------------------------------------------------------------
// f32 -> bf16 converters
// ---------------------------------------------------------------------------
__global__ void k_cvt(const float* __restrict__ s, unsigned short* __restrict__ d, int n) {
  for (int i = blockIdx.x * blockDim.x + threadIdx.x; i < n; i += gridDim.x * blockDim.x)
    d[i] = f2bf(s[i]);
}

// src [G][R][C] f32 -> dst [G][C][R] bf16 (store B-side matrices K-contiguous)
__global__ void k_cvt_t(const float* __restrict__ s, unsigned short* __restrict__ d,
                        int G, int R, int C) {
  int n = G * R * C;
  for (int i = blockIdx.x * blockDim.x + threadIdx.x; i < n; i += gridDim.x * blockDim.x) {
    int g = i / (R * C);
    int rem = i - g * R * C;
    int r = rem / C;
    int c = rem - r * C;
    d[(size_t)g * R * C + (size_t)c * R + r] = f2bf(s[i]);
  }
}

// ---------------------------------------------------------------------------
// Projection GEMMs: out = x @ W + b.  blockIdx.y = p: 0..15 q-heads,
// 16..31 k-heads, 32 = shared v (stored transposed [D][M] for later use).
// 64-wide K phases: 2 WMMAs per wave per barrier phase.
// ---------------------------------------------------------------------------
__global__ __launch_bounds__(128) void k_proj(
    const unsigned short* __restrict__ xb,
    const unsigned short* __restrict__ wqt, const unsigned short* __restrict__ wkt,
    const unsigned short* __restrict__ wvt,
    const float* __restrict__ bq, const float* __restrict__ bk, const float* __restrict__ bv,
    unsigned short* __restrict__ qo, unsigned short* __restrict__ ko,
    unsigned short* __restrict__ vt) {
  __shared__ unsigned short As[16][64];
  __shared__ unsigned short Bs[64][64];
  const int tid = threadIdx.x, lane = tid & 31, w = tid >> 5;
  const int m0 = blockIdx.x * 16;
  const int p = blockIdx.y;

  const unsigned short* Bsrc; const float* bias; unsigned short* outp; bool tr = false;
  if (p < NH_)      { Bsrc = wqt + (size_t)p * D_ * H_;        bias = bq + p * D_;        outp = qo + (size_t)p * M_ * D_; }
  else if (p < 32)  { Bsrc = wkt + (size_t)(p - 16) * D_ * H_; bias = bk + (p - 16) * D_; outp = ko + (size_t)(p - 16) * M_ * D_; }
  else              { Bsrc = wvt; bias = bv; outp = vt; tr = true; }

  v8f acc = {};
  for (int k0 = 0; k0 < H_; k0 += 64) {
    { int r = tid >> 3, cg = (tid & 7) * 8;                    // A: 16x64
      cp16(&As[r][cg], &xb[(size_t)(m0 + r) * H_ + k0 + cg]); }
    { int n = tid >> 1, kg = (tid & 1) * 32;                   // B^T: 64 N-rows x 64 K
      const unsigned short* src = &Bsrc[(size_t)n * H_ + k0 + kg];
      cp16(&Bs[n][kg],      src);
      cp16(&Bs[n][kg + 8],  src + 8);
      cp16(&Bs[n][kg + 16], src + 16);
      cp16(&Bs[n][kg + 24], src + 24); }
    async_join();
    __syncthreads();
#pragma unroll
    for (int kt = 0; kt < 2; ++kt) {
      v16bf a  = frag_a(&As[0][0], 64, lane, kt * 32);
      v16bf bf = frag_b(&Bs[0][0], 64, lane, w * 16, kt * 32);
      acc = WMMA_BF16(a, bf, acc);
    }
    __syncthreads();
  }
  int nn = w * 16 + (lane & 15);
  float bs = bias[nn];
#pragma unroll
  for (int i = 0; i < 8; ++i) {
    int m = m0 + ((lane >> 4) << 3) + i;
    unsigned short v = f2bf(acc[i] + bs);
    if (tr) outp[(size_t)nn * M_ + m] = v;       // v stored [D][M]
    else    outp[(size_t)m * D_ + nn] = v;       // q/k stored [M][D]
  }
}

// ---------------------------------------------------------------------------
// Attention: per (row-tile, batch) block; loop all 16 heads in-block,
// accumulating the head-mean of softmax weights straight into d_out.
// ---------------------------------------------------------------------------
__global__ __launch_bounds__(128) void k_attn(
    const unsigned short* __restrict__ q, const unsigned short* __restrict__ k,
    float* __restrict__ outw, unsigned short* __restrict__ wbar) {
  __shared__ float S[16][T_];                    // 128 KB score strip (f32)
  __shared__ unsigned short qs[16][64];
  __shared__ unsigned short ks[4][16][64];       // per-wave k tile
  __shared__ float red[16][8];
  __shared__ float rowmax[16];
  __shared__ float rowrcp[16];

  const int tid = threadIdx.x, lane = tid & 31, w = tid >> 5;
  const int mtile = blockIdx.x, b = blockIdx.y;
  const int m0 = mtile * 16, NS = mtile + 1, Tc = NS * 16;
  const size_t rowbase = (size_t)(b * T_ + m0);

  // masked upper-triangle columns are exact zeros in both outputs
  for (int r = 0; r < 16; ++r)
    for (int c = Tc + tid; c < T_; c += 128) {
      size_t g = (rowbase + r) * T_ + c;
      outw[g] = 0.0f;
      wbar[g] = 0;
    }

  for (int h = 0; h < NH_; ++h) {
    { int r = tid >> 3, cg = (tid & 7) * 8;      // stage q tile 16x64
      cp16(&qs[r][cg], &q[((size_t)h * M_ + rowbase + r) * D_ + cg]); }
    async_join();
    __syncthreads();

    for (int j = w; j < NS; j += 4) {            // s-tiles round-robin over waves
      int s0 = j * 16;
      { int r = lane & 15, hf = lane >> 4;       // stage k tile 16x64 (per wave)
        const unsigned short* src = &k[((size_t)h * M_ + (size_t)b * T_ + s0 + r) * D_ + hf * 32];
        unsigned short* dst = &ks[w][r][hf * 32];
        cp16(dst,      src);
        cp16(dst + 8,  src + 8);
        cp16(dst + 16, src + 16);
        cp16(dst + 24, src + 24); }
      async_join();                              // per-wave tile, no barrier needed
      v8f acc = {};
#pragma unroll
      for (int kt = 0; kt < 2; ++kt) {
        v16bf a  = frag_a(&qs[0][0], 64, lane, kt * 32);
        v16bf bf = frag_b(&ks[w][0][0], 64, lane, 0, kt * 32);  // k row-major == (k^T) K-contig
        acc = WMMA_BF16(a, bf, acc);
      }
#pragma unroll
      for (int i = 0; i < 8; ++i) {
        int rr = ((lane >> 4) << 3) + i;
        int ss = s0 + (lane & 15);
        float v = acc[i] * SCALE_;
        if (ss > m0 + rr) v = -__builtin_inff(); // causal mask
        S[rr][ss] = v;
      }
    }
    __syncthreads();

    { int r = tid >> 3, c0 = tid & 7;            // row max (8 partials/row)
      float m = -__builtin_inff();
      for (int s = c0; s < Tc; s += 8) m = fmaxf(m, S[r][s]);
      red[r][c0] = m; }
    __syncthreads();
    if (tid < 16) {
      float m = red[tid][0];
#pragma unroll
      for (int i = 1; i < 8; ++i) m = fmaxf(m, red[tid][i]);
      rowmax[tid] = m;
    }
    __syncthreads();
    { int r = tid >> 3, c0 = tid & 7;            // row sum of exp
      float rm = rowmax[r], sum = 0.f;
      for (int s = c0; s < Tc; s += 8) sum += __expf(S[r][s] - rm);
      red[r][c0] = sum; }
    __syncthreads();
    if (tid < 16) {
      float sum = 0.f;
#pragma unroll
      for (int i = 0; i < 8; ++i) sum += red[tid][i];
      rowrcp[tid] = 1.0f / sum;
    }
    __syncthreads();

    const float inh = 1.0f / (float)NH_;
    for (int r = 0; r < 16; ++r) {
      float rm = rowmax[r], rc = rowrcp[r];
      for (int c = tid; c < Tc; c += 128) {      // coalesced accumulate
        float pv = __expf(S[r][c] - rm) * rc * inh;
        size_t g = (rowbase + r) * T_ + c;
        float a2 = (h == 0) ? pv : (outw[g] + pv);
        outw[g] = a2;
        if (h == NH_ - 1) wbar[g] = f2bf(a2);    // bf16 copy for next GEMM
      }
    }
    __syncthreads();
  }
}

// ---------------------------------------------------------------------------
// mh = mean_weights @ v   (K truncated at the causal diagonal; 64-wide phases)
// ---------------------------------------------------------------------------
__global__ __launch_bounds__(128) void k_mh(
    const unsigned short* __restrict__ wbar, const unsigned short* __restrict__ vt,
    unsigned short* __restrict__ mh) {
  __shared__ unsigned short As[16][64];
  __shared__ unsigned short Bs[64][64];
  const int tid = threadIdx.x, lane = tid & 31, w = tid >> 5;
  const int mtile = blockIdx.x, b = blockIdx.y;
  const int m0 = mtile * 16;
  const int phases = (mtile + 4) >> 2;           // ceil((mtile+1)*16 / 64)
  const size_t rowbase = (size_t)(b * T_ + m0);

  v8f acc = {};
  for (int ph = 0; ph < phases; ++ph) {
    int k0 = ph * 64;
    { int r = tid >> 3, cg = (tid & 7) * 8;      // A: 16x64 (zeros past diagonal)
      cp16(&As[r][cg], &wbar[(rowbase + r) * T_ + k0 + cg]); }
    { int n = tid >> 1, kg = (tid & 1) * 32;     // vt is [D][M]: K contiguous
      const unsigned short* src = &vt[(size_t)n * M_ + (size_t)b * T_ + k0 + kg];
      cp16(&Bs[n][kg],      src);
      cp16(&Bs[n][kg + 8],  src + 8);
      cp16(&Bs[n][kg + 16], src + 16);
      cp16(&Bs[n][kg + 24], src + 24); }
    async_join();
    __syncthreads();
#pragma unroll
    for (int kt = 0; kt < 2; ++kt) {
      v16bf a  = frag_a(&As[0][0], 64, lane, kt * 32);
      v16bf bf = frag_b(&Bs[0][0], 64, lane, w * 16, kt * 32);
      acc = WMMA_BF16(a, bf, acc);
    }
    __syncthreads();
  }
  int nn = w * 16 + (lane & 15);
#pragma unroll
  for (int i = 0; i < 8; ++i) {
    size_t m = rowbase + ((lane >> 4) << 3) + i;
    mh[m * D_ + nn] = f2bf(acc[i]);
  }
}

// ---------------------------------------------------------------------------
// out = mh @ Wo + bo   (K = 64 -> 2 WMMA steps)
// ---------------------------------------------------------------------------
__global__ __launch_bounds__(128) void k_out(
    const unsigned short* __restrict__ mh, const unsigned short* __restrict__ wot,
    const float* __restrict__ bo, float* __restrict__ out) {
  __shared__ unsigned short As[16][64];
  __shared__ unsigned short Bs[64][64];
  const int tid = threadIdx.x, lane = tid & 31, w = tid >> 5;
  const int m0 = blockIdx.x * 16;
  const int nblk = blockIdx.y * 64;

  { int r = tid >> 3, cg = (tid & 7) * 8;
    cp16(&As[r][cg], &mh[(size_t)(m0 + r) * D_ + cg]); }
  { int n = tid >> 1, kg = (tid & 1) * 32;       // Wo^T is [H][D]: K contiguous
    const unsigned short* src = &wot[(size_t)(nblk + n) * D_ + kg];
    cp16(&Bs[n][kg],      src);
    cp16(&Bs[n][kg + 8],  src + 8);
    cp16(&Bs[n][kg + 16], src + 16);
    cp16(&Bs[n][kg + 24], src + 24); }
  async_join();
  __syncthreads();

  v8f acc = {};
#pragma unroll
  for (int kt = 0; kt < 2; ++kt) {
    v16bf a  = frag_a(&As[0][0], 64, lane, kt * 32);
    v16bf bf = frag_b(&Bs[0][0], 64, lane, w * 16, kt * 32);
    acc = WMMA_BF16(a, bf, acc);
  }
  int hcol = nblk + w * 16 + (lane & 15);
  float bias = bo[hcol];
#pragma unroll
  for (int i = 0; i < 8; ++i) {
    int m = m0 + ((lane >> 4) << 3) + i;
    out[(size_t)m * H_ + hcol] = acc[i] + bias;
  }
}

// ---------------------------------------------------------------------------
extern "C" void kernel_launch(void* const* d_in, const int* in_sizes, int n_in,
                              void* d_out, int out_size, void* d_ws, size_t ws_size,
                              hipStream_t stream) {
  (void)in_sizes; (void)n_in; (void)out_size; (void)ws_size;
  const float* x  = (const float*)d_in[0];
  const float* Wq = (const float*)d_in[1];
  const float* bq = (const float*)d_in[2];
  const float* Wk = (const float*)d_in[3];
  const float* bk = (const float*)d_in[4];
  const float* Wv = (const float*)d_in[5];
  const float* bv = (const float*)d_in[6];
  const float* Wo = (const float*)d_in[7];
  const float* bo = (const float*)d_in[8];
  float* out = (float*)d_out;                    // [B,T,H] then [B,T,T]

  unsigned short* ws = (unsigned short*)d_ws;
  size_t off = 0;
  unsigned short* xb   = ws + off; off += (size_t)M_ * H_;        // x bf16
  unsigned short* wqt  = ws + off; off += (size_t)NH_ * H_ * D_;  // Wq^T per head [D][H]
  unsigned short* wkt  = ws + off; off += (size_t)NH_ * H_ * D_;
  unsigned short* wvt  = ws + off; off += (size_t)H_ * D_;        // Wv^T [D][H]
  unsigned short* wot  = ws + off; off += (size_t)D_ * H_;        // Wo^T [H][D]
  unsigned short* qo   = ws + off; off += (size_t)NH_ * M_ * D_;  // q [NH][M][D]
  unsigned short* ko   = ws + off; off += (size_t)NH_ * M_ * D_;  // k [NH][M][D]
  unsigned short* vt   = ws + off; off += (size_t)M_ * D_;        // v^T [D][M]
  unsigned short* wbar = ws + off; off += (size_t)B_ * T_ * T_;   // mean weights bf16
  unsigned short* mh   = ws + off; off += (size_t)M_ * D_;        // mean head [M][D]

  k_cvt  <<<1024, 256, 0, stream>>>(x, xb, M_ * H_);
  k_cvt_t<<<1024, 256, 0, stream>>>(Wq, wqt, NH_, H_, D_);
  k_cvt_t<<<1024, 256, 0, stream>>>(Wk, wkt, NH_, H_, D_);
  k_cvt_t<<<256,  256, 0, stream>>>(Wv, wvt, 1, H_, D_);
  k_cvt_t<<<256,  256, 0, stream>>>(Wo, wot, 1, D_, H_);

  k_proj<<<dim3(M_ / 16, 33), 128, 0, stream>>>(xb, wqt, wkt, wvt, bq, bk, bv, qo, ko, vt);

  float* outw = out + (size_t)B_ * T_ * H_;      // mean_weights region of d_out
  k_attn<<<dim3(T_ / 16, B_), 128, 0, stream>>>(qo, ko, outw, wbar);

  k_mh <<<dim3(T_ / 16, B_), 128, 0, stream>>>(wbar, vt, mh);
  k_out<<<dim3(M_ / 16, H_ / 64), 128, 0, stream>>>(mh, wot, bo, out);
}
